// GAT_60997125538330
// MI455X (gfx1250) — matrix operand order repaired and checked by reading, not compiled, not run on hardware
//
#include <hip/hip_runtime.h>
#include <hip/hip_bf16.h>
#include <stdint.h>

#define N_NODES 50000
#define N_EDGES 400000
#define E_TOT   450000   /* edges + self loops */
#define F_IN    128
#define HID     64
#define HEADS   8
#define HCDIM   512      /* HEADS*HID */
#define NCLS    16

typedef __attribute__((ext_vector_type(16))) _Float16 v16h;
typedef __attribute__((ext_vector_type(8)))  _Float16 v8h;
typedef __attribute__((ext_vector_type(8)))  float    v8f;

// ---------------- utility kernels ----------------

__global__ void k_fill_u32(uint32_t* __restrict__ p, uint32_t v, long long n) {
  long long i  = (long long)blockIdx.x * blockDim.x + threadIdx.x;
  long long st = (long long)gridDim.x * blockDim.x;
  for (; i < n; i += st) p[i] = v;
}

__global__ void k_cast_f16(const float* __restrict__ x, _Float16* __restrict__ y, long long n) {
  long long i  = (long long)blockIdx.x * blockDim.x + threadIdx.x;
  long long st = (long long)gridDim.x * blockDim.x;
  for (; i < n; i += st) y[i] = (_Float16)x[i];
}

// W: [K,N] row-major f32  ->  Wt: [N,K] f16
__global__ void k_transpose_cast(const float* __restrict__ W, _Float16* __restrict__ Wt,
                                 int K, int N) {
  long long n  = (long long)K * N;
  long long i  = (long long)blockIdx.x * blockDim.x + threadIdx.x;
  long long st = (long long)gridDim.x * blockDim.x;
  for (; i < n; i += st) {
    int k = (int)(i / N), c = (int)(i % N);
    Wt[(size_t)c * K + k] = (_Float16)W[i];
  }
}

// ---------------- WMMA GEMM: C[M,N] = A[M,K](f16) * Bt[N,K](f16), f32 accum ----------------
// One wave per 16 x (16*NG) strip: A fragment loaded once per k-step, reused NG times.
// All loads of a k-step are issued before any WMMA so waits can be partial.
// M%16==0, N%(16*NG)==0, K%32==0.
template <int NG>
__global__ void k_wmma_gemm(const _Float16* __restrict__ A, const _Float16* __restrict__ Bt,
                            float* __restrict__ C, int M, int N, int K) {
  int wavesPerBlk = blockDim.x >> 5;
  int tile = blockIdx.x * wavesPerBlk + (threadIdx.x >> 5);
  int tilesN = N / (16 * NG);
  int totalTiles = (M >> 4) * tilesN;
  if (tile >= totalTiles) return;
  int tm = tile / tilesN, tn = tile % tilesN;

  int lane = threadIdx.x & 31;
  int hl   = lane >> 4;       // 0: lanes 0-15, 1: lanes 16-31
  int lr   = lane & 15;

  const _Float16* arow = A + (size_t)(tm * 16 + lr) * K;
  const _Float16* brow[NG];
  #pragma unroll
  for (int j = 0; j < NG; ++j)
    brow[j] = Bt + (size_t)((tn * NG + j) * 16 + lr) * K;

  v8f acc[NG];
  #pragma unroll
  for (int j = 0; j < NG; ++j) acc[j] = (v8f){};

  for (int k0 = 0; k0 < K; k0 += 32) {
    // Stage ALL loads for this k-step first (2xA + 2*NG xB b128 loads),
    // so the compiler can use partial s_wait_loadcnt and overlap with WMMA.
    // A 16x32 layout: lane half 0 -> K {0..7, 16..23}; half 1 -> K {8..15, 24..31}
    v8h a0 = *(const v8h*)(arow + k0 + hl * 8);
    v8h a1 = *(const v8h*)(arow + k0 + 16 + hl * 8);
    v8h b0[NG], b1[NG];
    #pragma unroll
    for (int j = 0; j < NG; ++j) {
      // B 32x16 layout (column = lr): half 0 -> K {0..7, 16..23}; half 1 -> K {8..15, 24..31}
      b0[j] = *(const v8h*)(brow[j] + k0 + hl * 8);
      b1[j] = *(const v8h*)(brow[j] + k0 + 16 + hl * 8);
    }
    v16h af = __builtin_shufflevector(a0, a1, 0,1,2,3,4,5,6,7,8,9,10,11,12,13,14,15);
    #pragma unroll
    for (int j = 0; j < NG; ++j) {
      v16h bf = __builtin_shufflevector(b0[j], b1[j],
                                        0,1,2,3,4,5,6,7,8,9,10,11,12,13,14,15);
      acc[j] = __builtin_amdgcn_wmma_f32_16x16x32_f16(false, af, false, bf,
                                                      (short)0, acc[j], false, false);
    }
  }
  // D layout: VGPR r -> row r (lanes 0-15) / row r+8 (lanes 16-31), col = lr
  #pragma unroll
  for (int j = 0; j < NG; ++j) {
    float* crow = C + (size_t)(tm * 16 + hl * 8) * N + (tn * NG + j) * 16 + lr;
    #pragma unroll
    for (int r = 0; r < 8; ++r) crow[(size_t)r * N] = acc[j][r];
  }
}

// ---------------- attention ----------------

// al_s[n,h] = dot(H[n,h,:], a_s[h,:]);  al_d likewise
__global__ void k_attn_logits(const float* __restrict__ Hm, const float* __restrict__ a_s,
                              const float* __restrict__ a_d, float* __restrict__ als,
                              float* __restrict__ ald, int nH, int C) {
  long long n  = (long long)N_NODES * nH;
  long long i  = (long long)blockIdx.x * blockDim.x + threadIdx.x;
  long long st = (long long)gridDim.x * blockDim.x;
  for (; i < n; i += st) {
    int node = (int)(i / nH), h = (int)(i % nH);
    const float* hp = Hm + (size_t)node * nH * C + (size_t)h * C;
    const float* sp = a_s + (size_t)h * C;
    const float* dp = a_d + (size_t)h * C;
    float s = 0.f, d = 0.f;
    for (int c = 0; c < C; ++c) { s += hp[c] * sp[c]; d += hp[c] * dp[c]; }
    als[i] = s; ald[i] = d;
  }
}

__device__ __forceinline__ void edge_sd(const int* __restrict__ ei, long long e,
                                        int& s, int& d) {
  if (e < N_EDGES) { s = ei[e]; d = ei[N_EDGES + e]; }
  else             { s = d = (int)(e - N_EDGES); }   // self loops
}
__device__ __forceinline__ unsigned okey(float f) {
  int i = __float_as_int(f);
  return (i >= 0) ? ((unsigned)i ^ 0x80000000u) : ~(unsigned)i;
}
__device__ __forceinline__ float odec(unsigned u) {
  int b = (u & 0x80000000u) ? (int)(u ^ 0x80000000u) : (int)(~u);
  return __int_as_float(b);
}
__device__ __forceinline__ float lrelu(float v) { return v > 0.f ? v : 0.2f * v; }

__global__ void k_edge_max(const int* __restrict__ ei, const float* __restrict__ als,
                           const float* __restrict__ ald, unsigned* __restrict__ m, int nH) {
  long long n  = (long long)E_TOT * nH;
  long long i  = (long long)blockIdx.x * blockDim.x + threadIdx.x;
  long long st = (long long)gridDim.x * blockDim.x;
  for (; i < n; i += st) {
    long long e = i / nH; int h = (int)(i % nH);
    int s, d; edge_sd(ei, e, s, d);
    float v = lrelu(als[(size_t)s * nH + h] + ald[(size_t)d * nH + h]);
    atomicMax(m + (size_t)d * nH + h, okey(v));
  }
}

__global__ void k_edge_expsum(const int* __restrict__ ei, const float* __restrict__ als,
                              const float* __restrict__ ald, const unsigned* __restrict__ m,
                              float* __restrict__ denom, float* __restrict__ exbuf, int nH) {
  long long n  = (long long)E_TOT * nH;
  long long i  = (long long)blockIdx.x * blockDim.x + threadIdx.x;
  long long st = (long long)gridDim.x * blockDim.x;
  for (; i < n; i += st) {
    long long e = i / nH; int h = (int)(i % nH);
    int s, d; edge_sd(ei, e, s, d);
    float v  = lrelu(als[(size_t)s * nH + h] + ald[(size_t)d * nH + h]);
    float ex = __expf(v - odec(m[(size_t)d * nH + h]));
    exbuf[i] = ex;
    atomicAdd(denom + (size_t)d * nH + h, ex);
  }
}

// out[dst,:] += alpha * H[src,:], one thread per (edge, 4-channel group)
__global__ void k_edge_aggregate(const int* __restrict__ ei, const float* __restrict__ Hm,
                                 const float* __restrict__ exbuf, const float* __restrict__ denom,
                                 float* __restrict__ out, int nH, int C) {
  int hcq = (nH * C) >> 2;
  long long n  = (long long)E_TOT * hcq;
  long long i  = (long long)blockIdx.x * blockDim.x + threadIdx.x;
  long long st = (long long)gridDim.x * blockDim.x;
  for (; i < n; i += st) {
    long long e = i / hcq; int q = (int)(i % hcq);
    int c0 = q << 2, h = c0 / C;
    int s, d; edge_sd(ei, e, s, d);
    float alpha = exbuf[(size_t)e * nH + h] / denom[(size_t)d * nH + h];
    float4 hv = *(const float4*)(Hm + (size_t)s * nH * C + c0);
    float* op = out + (size_t)d * nH * C + c0;
    atomicAdd(op + 0, alpha * hv.x);
    atomicAdd(op + 1, alpha * hv.y);
    atomicAdd(op + 2, alpha * hv.z);
    atomicAdd(op + 3, alpha * hv.w);
  }
}

// y_f16 = elu(x + bias[i % ncol])  (fused bias+ELU+cast for next layer's GEMM input)
__global__ void k_bias_elu_cast(const float* __restrict__ x, const float* __restrict__ b,
                                _Float16* __restrict__ y, long long n, int ncol) {
  long long i  = (long long)blockIdx.x * blockDim.x + threadIdx.x;
  long long st = (long long)gridDim.x * blockDim.x;
  for (; i < n; i += st) {
    float v = x[i] + b[i % ncol];
    v = v > 0.f ? v : (__expf(v) - 1.f);
    y[i] = (_Float16)v;
  }
}

__global__ void k_bias_add(float* __restrict__ x, const float* __restrict__ b,
                           long long n, int ncol) {
  long long i  = (long long)blockIdx.x * blockDim.x + threadIdx.x;
  long long st = (long long)gridDim.x * blockDim.x;
  for (; i < n; i += st) x[i] += b[i % ncol];
}

// ---------------- host ----------------

static inline int gs_blocks(long long n) {
  long long b = (n + 255) / 256;
  return (int)(b > 16384 ? 16384 : (b < 1 ? 1 : b));
}

extern "C" void kernel_launch(void* const* d_in, const int* in_sizes, int n_in,
                              void* d_out, int out_size, void* d_ws, size_t ws_size,
                              hipStream_t stream) {
  const float* x    = (const float*)d_in[0];
  const int*   ei   = (const int*)  d_in[1];
  const float* W[3]   = {(const float*)d_in[2], (const float*)d_in[6], (const float*)d_in[10]};
  const float* a_s[3] = {(const float*)d_in[3], (const float*)d_in[7], (const float*)d_in[11]};
  const float* a_d[3] = {(const float*)d_in[4], (const float*)d_in[8], (const float*)d_in[12]};
  const float* bia[3] = {(const float*)d_in[5], (const float*)d_in[9], (const float*)d_in[13]};

  // workspace carve (deterministic, same offsets every call)
  char* p = (char*)d_ws;
  auto carve = [&](size_t bytes) { void* r = (void*)p; p += (bytes + 255) & ~(size_t)255; return r; };
  _Float16* Xh    = (_Float16*)carve((size_t)N_NODES * HCDIM * 2);
  _Float16* Wt    = (_Float16*)carve((size_t)HCDIM * HCDIM * 2);
  float*    bufA  = (float*)   carve((size_t)N_NODES * HCDIM * 4);   // GEMM out (h)
  float*    bufB  = (float*)   carve((size_t)N_NODES * HCDIM * 4);   // aggregated out
  float*    als   = (float*)   carve((size_t)N_NODES * HEADS * 4);
  float*    ald   = (float*)   carve((size_t)N_NODES * HEADS * 4);
  unsigned* mbuf  = (unsigned*)carve((size_t)N_NODES * HEADS * 4);
  float*    denom = (float*)   carve((size_t)N_NODES * HEADS * 4);
  float*    exbuf = (float*)   carve((size_t)E_TOT   * HEADS * 4);

  const unsigned NEG_INF_KEY = 0x007FFFFFu;  // okey(-inf)

  // layer-0 input cast: [N_NODES, F_IN] f32 -> f16
  k_cast_f16<<<gs_blocks((long long)N_NODES * F_IN), 256, 0, stream>>>(
      x, Xh, (long long)N_NODES * F_IN);

  const int Kdim[3] = {F_IN, HCDIM, HCDIM};
  const int Ndim[3] = {HCDIM, HCDIM, NCLS};
  const int nHL[3]  = {HEADS, HEADS, 1};
  const int CL[3]   = {HID, HID, NCLS};

  for (int L = 0; L < 3; ++L) {
    const int K = Kdim[L], N = Ndim[L], nH = nHL[L], C = CL[L];

    // weights -> f16 [N,K]
    k_transpose_cast<<<gs_blocks((long long)K * N), 256, 0, stream>>>(W[L], Wt, K, N);

    // h = X @ W  (WMMA).  512-wide layers: 16x64 strip per wave (NG=4); classifier: NG=1.
    if (N == HCDIM) {
      int tiles = (N_NODES / 16) * (N / 64);
      k_wmma_gemm<4><<<(tiles + 7) / 8, 256, 0, stream>>>(Xh, Wt, bufA, N_NODES, N, K);
    } else {
      int tiles = (N_NODES / 16) * (N / 16);
      k_wmma_gemm<1><<<(tiles + 7) / 8, 256, 0, stream>>>(Xh, Wt, bufA, N_NODES, N, K);
    }

    // attention logits per (node, head)
    k_attn_logits<<<gs_blocks((long long)N_NODES * nH), 256, 0, stream>>>(
        bufA, a_s[L], a_d[L], als, ald, nH, C);

    // softmax state init
    k_fill_u32<<<gs_blocks((long long)N_NODES * nH), 256, 0, stream>>>(
        mbuf, NEG_INF_KEY, (long long)N_NODES * nH);
    k_fill_u32<<<gs_blocks((long long)N_NODES * nH), 256, 0, stream>>>(
        (uint32_t*)denom, 0u, (long long)N_NODES * nH);

    k_edge_max<<<gs_blocks((long long)E_TOT * nH), 256, 0, stream>>>(ei, als, ald, mbuf, nH);
    k_edge_expsum<<<gs_blocks((long long)E_TOT * nH), 256, 0, stream>>>(
        ei, als, ald, mbuf, denom, exbuf, nH);

    // aggregate
    float* outp = (L == 2) ? (float*)d_out : bufB;
    long long outElems = (long long)N_NODES * nH * C;
    k_fill_u32<<<gs_blocks(outElems), 256, 0, stream>>>((uint32_t*)outp, 0u, outElems);
    k_edge_aggregate<<<gs_blocks((long long)E_TOT * ((nH * C) >> 2)), 256, 0, stream>>>(
        ei, bufA, exbuf, denom, outp, nH, C);

    if (L < 2) {
      // bias + ELU, cast to f16 as next layer's GEMM input
      k_bias_elu_cast<<<gs_blocks(outElems), 256, 0, stream>>>(
          bufB, bia[L], Xh, outElems, nH * C);
    } else {
      // final: heads=1 mean is identity; just add bias
      k_bias_add<<<gs_blocks(outElems), 256, 0, stream>>>(
          (float*)d_out, bia[L], outElems, NCLS);
    }
  }
}
